// SelfAttention_39444979646720
// MI455X (gfx1250) — matrix operand order repaired and checked by reading, not compiled, not run on hardware
//
#include <hip/hip_runtime.h>
#include <hip/hip_bf16.h>

typedef _Float16 f16;
typedef __attribute__((ext_vector_type(16))) _Float16 v16h;
typedef __attribute__((ext_vector_type(8)))  _Float16 v8h;
typedef __attribute__((ext_vector_type(8)))  float    v8f;

#define BB 2
#define SS 2048
#define DD 768
#define HH 12
#define DH 64

// ---------------- fragment helpers (layouts per CDNA5 ISA) ----------------

__device__ __forceinline__ v16h cat16(v8h lo, v8h hi) {
  return __builtin_shufflevector(lo, hi, 0,1,2,3,4,5,6,7,8,9,10,11,12,13,14,15);
}

// A-matrix 16x32 f16: lane m=M (both halves), half i -> K = (i/8)*16 + lh*8 + (i%8)
__device__ __forceinline__ v16h load_afrag(const f16* base, int ld, int row0, int k0, int lane) {
  int m = lane & 15, lh = lane >> 4;
  const f16* rp = base + (size_t)(row0 + m) * ld + k0 + lh * 8;
  v8h lo = *(const v8h*)rp;          // K = k0 + lh*8 + 0..7
  v8h hi = *(const v8h*)(rp + 16);   // K = k0 + 16 + lh*8 + 0..7
  return cat16(lo, hi);
}

// B-matrix 32x16 f16 from "transposed" storage Bt[n][k] (ld = K stride of rows):
// lane m = N, half i -> K = (i/8)*16 + lh*8 + (i%8)
__device__ __forceinline__ v16h load_bfrag(const f16* baseT, int ld, int n0, int k0, int lane) {
  int m = lane & 15, lh = lane >> 4;
  const f16* rp = baseT + (size_t)(n0 + m) * ld + k0 + lh * 8;
  v8h lo = *(const v8h*)rp;
  v8h hi = *(const v8h*)(rp + 16);
  return cat16(lo, hi);
}

__device__ __forceinline__ v8f wmma32(v16h a, v16h b, v8f c) {
  return __builtin_amdgcn_wmma_f32_16x16x32_f16(false, a, false, b, (short)0, c, false, false);
}

// ---------------- prep kernels ----------------

__global__ void convert_x_kernel(const float* __restrict__ x, f16* __restrict__ xh, int n) {
  int i = blockIdx.x * blockDim.x + threadIdx.x;
  if (i < n) xh[i] = (f16)x[i];
}

__global__ void transpose_w_kernel(const float* __restrict__ Wq, const float* __restrict__ Wk,
                                   const float* __restrict__ Wv, const float* __restrict__ Wo,
                                   f16* __restrict__ Wqt, f16* __restrict__ Wkt,
                                   f16* __restrict__ Wvt, f16* __restrict__ Wot) {
  int idx = blockIdx.x * blockDim.x + threadIdx.x;           // 0 .. 768*768-1
  int which = blockIdx.y;
  const float* src = which == 0 ? Wq : which == 1 ? Wk : which == 2 ? Wv : Wo;
  f16*         dst = which == 0 ? Wqt : which == 1 ? Wkt : which == 2 ? Wvt : Wot;
  int n = idx / DD, k = idx % DD;
  dst[(size_t)n * DD + k] = (f16)src[(size_t)k * DD + n];    // Wt[n][k] = W[k][n]
}

// ---------------- fused QKV projection ----------------
// One wave computes a 32x64 tile of one of Q/K/V (2 M-tiles share each B fragment).
// M = B*S = 4096, N = 768, K = 768.
__global__ void __launch_bounds__(128) qkv_gemm_kernel(
    const f16* __restrict__ xh,
    const f16* __restrict__ Wqt, const f16* __restrict__ Wkt, const f16* __restrict__ Wvt,
    const float* __restrict__ bq, const float* __restrict__ bk, const float* __restrict__ bv,
    f16* __restrict__ Qh, f16* __restrict__ Kh, f16* __restrict__ Vt) {
  int lane = threadIdx.x & 31;
  int wid  = threadIdx.x >> 5;
  int w = blockIdx.x * 4 + wid;                 // 0 .. 4607
  int which = w / (128 * 12);                   // 0=Q 1=K 2=V
  int rem   = w % (128 * 12);
  int mt = rem % 128;                           // 32-row tile
  int ng = rem / 128;                           // 0..11 (64 output cols each)
  const f16*   Wt   = which == 0 ? Wqt : which == 1 ? Wkt : Wvt;
  const float* bias = which == 0 ? bq  : which == 1 ? bk  : bv;
  int m = lane & 15, lh = lane >> 4;
  int row0 = mt * 32;

  v8f acc0[4] = {v8f{}, v8f{}, v8f{}, v8f{}};
  v8f acc1[4] = {v8f{}, v8f{}, v8f{}, v8f{}};
  for (int ks = 0; ks < 24; ++ks) {
    int k0 = ks * 32;
    v16h a0 = load_afrag(xh, DD, row0,      k0, lane);
    v16h a1 = load_afrag(xh, DD, row0 + 16, k0, lane);
    #pragma unroll
    for (int j = 0; j < 4; ++j) {
      v16h b = load_bfrag(Wt, DD, ng * 64 + j * 16, k0, lane);
      acc0[j] = wmma32(a0, b, acc0[j]);
      acc1[j] = wmma32(a1, b, acc1[j]);
    }
  }

  int b = row0 >> 11;                           // batch (32 | 2048 so no straddle)
  #pragma unroll
  for (int t = 0; t < 2; ++t) {
    int sb = ((row0 & 2047) + t * 16) + lh * 8; // first seq row this lane-half owns
    #pragma unroll
    for (int j = 0; j < 4; ++j) {
      const v8f& ac = t == 0 ? acc0[j] : acc1[j];
      int n = ng * 64 + j * 16 + m;             // global output column
      int h = n >> 6, dh = n & 63;
      float bi = bias[n];
      if (which == 2) {
        // V transposed: Vt[((b*H+h)*64+dh)*2048 + s], 8 consecutive s -> vector store
        v8h pk;
        #pragma unroll
        for (int r = 0; r < 8; ++r) pk[r] = (f16)(ac[r] + bi);
        *(v8h*)(Vt + (((size_t)b * HH + h) * DH + dh) * SS + sb) = pk;
      } else {
        f16* dst = (which == 0) ? Qh : Kh;
        float sc = (which == 0) ? 0.125f : 1.0f;  // fold 1/sqrt(Dh) into Q
        #pragma unroll
        for (int r = 0; r < 8; ++r) {
          int s = sb + r;
          dst[(((size_t)b * HH + h) * SS + s) * DH + dh] = (f16)((ac[r] + bi) * sc);
        }
      }
    }
  }
}

// ---------------- flash attention ----------------
// One wave per (b, h, 16-query tile).  32-key chunks, online softmax.
__global__ void __launch_bounds__(128) attn_kernel(
    const f16* __restrict__ Qh, const f16* __restrict__ Kh, const f16* __restrict__ Vt,
    const unsigned char* __restrict__ kpm, f16* __restrict__ attnout) {
  __shared__ __align__(16) f16 ldsP[4][16 * 32];
  int lane = threadIdx.x & 31;
  int wid  = threadIdx.x >> 5;
  int w = blockIdx.x * 4 + wid;                 // 0 .. 3071
  int qt = w & 127;
  int h  = (w >> 7) % 12;
  int b  = w / (128 * 12);
  int q0 = qt * 16;
  int m = lane & 15, lh = lane >> 4;

  const f16* Qb = Qh + ((size_t)b * HH + h) * SS * DH;
  const f16* Kb = Kh + ((size_t)b * HH + h) * SS * DH;
  const f16* Vb = Vt + ((size_t)b * HH + h) * DH * SS;
  f16* P = ldsP[wid];

  v16h aq0 = load_afrag(Qb, DH, q0, 0, lane);
  v16h aq1 = load_afrag(Qb, DH, q0, 32, lane);

  v8f acc[4] = {v8f{}, v8f{}, v8f{}, v8f{}};
  float mrun[8], lrun[8];
  #pragma unroll
  for (int r = 0; r < 8; ++r) { mrun[r] = -3.0e38f; lrun[r] = 0.0f; }

  int nch = (q0 + 16 + 31) >> 5;                // chunks of 32 keys, keys < q0+16 needed
  for (int c = 0; c < nch; ++c) {
    int k0 = c << 5;
    // ---- scores: two 16x16 tiles (keys k0..k0+15, k0+16..k0+31), Dh=64 = 2 K-steps
    v8f sc0 = {}, sc1 = {};
    {
      v16h bk0 = load_bfrag(Kb, DH, k0,      0, lane);
      v16h bk1 = load_bfrag(Kb, DH, k0,     32, lane);
      sc0 = wmma32(aq0, bk0, sc0);
      sc0 = wmma32(aq1, bk1, sc0);
      v16h bk2 = load_bfrag(Kb, DH, k0 + 16,  0, lane);
      v16h bk3 = load_bfrag(Kb, DH, k0 + 16, 32, lane);
      sc1 = wmma32(aq0, bk2, sc1);
      sc1 = wmma32(aq1, bk3, sc1);
    }
    // ---- masks (this lane's key column for each sub-tile)
    int key0 = k0 + m, key1 = k0 + 16 + m;
    bool pad0 = kpm[(size_t)b * SS + key0] != 0;
    bool pad1 = kpm[(size_t)b * SS + key1] != 0;
    // ---- online softmax per row (row = lh*8 + r lives entirely in this half-wave)
    #pragma unroll
    for (int r = 0; r < 8; ++r) {
      int q = q0 + lh * 8 + r;
      float v0 = sc0[r]; if (key0 > q) v0 += -1.0e9f; if (pad0) v0 = -3.0e38f;
      float v1 = sc1[r]; if (key1 > q) v1 += -1.0e9f; if (pad1) v1 = -3.0e38f;
      float tmax = fmaxf(v0, v1);
      tmax = fmaxf(tmax, __shfl_xor(tmax, 1, 32));
      tmax = fmaxf(tmax, __shfl_xor(tmax, 2, 32));
      tmax = fmaxf(tmax, __shfl_xor(tmax, 4, 32));
      tmax = fmaxf(tmax, __shfl_xor(tmax, 8, 32));
      float mnew  = fmaxf(mrun[r], tmax);
      float scale = __expf(mrun[r] - mnew);
      float p0 = __expf(v0 - mnew);
      float p1 = __expf(v1 - mnew);
      float rs = p0 + p1;
      rs += __shfl_xor(rs, 1, 32);
      rs += __shfl_xor(rs, 2, 32);
      rs += __shfl_xor(rs, 4, 32);
      rs += __shfl_xor(rs, 8, 32);
      lrun[r] = lrun[r] * scale + rs;
      mrun[r] = mnew;
      acc[0][r] *= scale; acc[1][r] *= scale; acc[2][r] *= scale; acc[3][r] *= scale;
      int row = lh * 8 + r;
      P[row * 32 + m]      = (f16)p0;
      P[row * 32 + 16 + m] = (f16)p1;
    }
    asm volatile("s_wait_dscnt 0" ::: "memory");
    // ---- P @ V : A = P (16x32 from LDS), B = Vt rows (dh contiguous over keys)
    v16h ap = load_afrag(P, 32, 0, 0, lane);
    #pragma unroll
    for (int j = 0; j < 4; ++j) {
      v16h bv = load_bfrag(Vb, SS, j * 16, k0, lane);
      acc[j] = wmma32(ap, bv, acc[j]);
    }
  }
  // ---- epilogue: normalize, zero padded query rows, write [B,S,D] f16
  #pragma unroll
  for (int r = 0; r < 8; ++r) {
    int q = q0 + lh * 8 + r;
    float inv = 1.0f / lrun[r];
    if (kpm[(size_t)b * SS + q] != 0) inv = 0.0f;
    size_t rowoff = ((size_t)b * SS + q) * DD + h * DH;
    #pragma unroll
    for (int j = 0; j < 4; ++j)
      attnout[rowoff + j * 16 + m] = (f16)(acc[j][r] * inv);
  }
}

// ---------------- output projection ----------------
// One wave computes a 32x64 tile of out = attnout @ WoT + bo (f32 result).
__global__ void __launch_bounds__(128) outproj_kernel(
    const f16* __restrict__ ah, const f16* __restrict__ Wot,
    const float* __restrict__ bo, float* __restrict__ out) {
  int lane = threadIdx.x & 31;
  int wid  = threadIdx.x >> 5;
  int w = blockIdx.x * 4 + wid;                 // 0 .. 1535
  int mt = w % 128;
  int ng = w / 128;
  int m = lane & 15, lh = lane >> 4;
  int row0 = mt * 32;

  v8f acc0[4] = {v8f{}, v8f{}, v8f{}, v8f{}};
  v8f acc1[4] = {v8f{}, v8f{}, v8f{}, v8f{}};
  for (int ks = 0; ks < 24; ++ks) {
    int k0 = ks * 32;
    v16h a0 = load_afrag(ah, DD, row0,      k0, lane);
    v16h a1 = load_afrag(ah, DD, row0 + 16, k0, lane);
    #pragma unroll
    for (int j = 0; j < 4; ++j) {
      v16h b = load_bfrag(Wot, DD, ng * 64 + j * 16, k0, lane);
      acc0[j] = wmma32(a0, b, acc0[j]);
      acc1[j] = wmma32(a1, b, acc1[j]);
    }
  }
  #pragma unroll
  for (int t = 0; t < 2; ++t) {
    #pragma unroll
    for (int j = 0; j < 4; ++j) {
      const v8f& ac = t == 0 ? acc0[j] : acc1[j];
      int n = ng * 64 + j * 16 + m;
      float bi = bo[n];
      #pragma unroll
      for (int r = 0; r < 8; ++r) {
        int gm = row0 + t * 16 + lh * 8 + r;
        out[(size_t)gm * DD + n] = ac[r] + bi;
      }
    }
  }
}

// ---------------- host launcher ----------------
extern "C" void kernel_launch(void* const* d_in, const int* in_sizes, int n_in,
                              void* d_out, int out_size, void* d_ws, size_t ws_size,
                              hipStream_t stream) {
  const float* x  = (const float*)d_in[0];
  const float* Wq = (const float*)d_in[1];
  const float* bq = (const float*)d_in[2];
  const float* Wk = (const float*)d_in[3];
  const float* bk = (const float*)d_in[4];
  const float* Wv = (const float*)d_in[5];
  const float* bv = (const float*)d_in[6];
  const float* Wo = (const float*)d_in[7];
  const float* bo = (const float*)d_in[8];
  // d_in[9] = causal_mask (recomputed arithmetically on device)
  const unsigned char* kpm = (const unsigned char*)d_in[10];

  char* ws = (char*)d_ws;
  const size_t XH  = 0;                                 // 4096*768*2      = 6,291,456
  const size_t WQT = 6291456;                           // 768*768*2 each
  const size_t WKT = WQT + 1179648;
  const size_t WVT = WKT + 1179648;
  const size_t WOT = WVT + 1179648;
  const size_t QH  = WOT + 1179648;                     // 11,010,048
  const size_t KH  = QH + 6291456;
  const size_t VT  = KH + 6291456;                      // end = 29,884,416 bytes

  f16* xh  = (f16*)(ws + XH);
  f16* Wqt = (f16*)(ws + WQT);
  f16* Wkt = (f16*)(ws + WKT);
  f16* Wvt = (f16*)(ws + WVT);
  f16* Wot = (f16*)(ws + WOT);
  f16* Qh  = (f16*)(ws + QH);
  f16* Kh  = (f16*)(ws + KH);
  f16* Vt  = (f16*)(ws + VT);
  f16* attnout = xh;   // xh fully consumed by qkv_gemm before attn writes here

  convert_x_kernel<<<12288, 256, 0, stream>>>(x, xh, BB * SS * DD);
  transpose_w_kernel<<<dim3(2304, 4), 256, 0, stream>>>(Wq, Wk, Wv, Wo, Wqt, Wkt, Wvt, Wot);
  qkv_gemm_kernel<<<1152, 128, 0, stream>>>(xh, Wqt, Wkt, Wvt, bq, bk, bv, Qh, Kh, Vt);
  attn_kernel<<<768, 128, 0, stream>>>(Qh, Kh, Vt, kpm, attnout);
  outproj_kernel<<<384, 128, 0, stream>>>(attnout, Wot, bo, (float*)d_out);
}
